// SoftDecisionTreeLayer_49709951484312
// MI455X (gfx1250) — matrix operand order repaired and checked by reading, not compiled, not run on hardware
//
#include <hip/hip_runtime.h>
#include <hip/hip_bf16.h>

// ---------- types matching gfx1250 WMMA builtin signatures ----------
typedef __attribute__((ext_vector_type(16))) __bf16       v16bf;
typedef __attribute__((ext_vector_type(8)))  float        v8f;
typedef __attribute__((ext_vector_type(8)))  unsigned int v8u;
typedef __attribute__((ext_vector_type(4)))  unsigned int v4u;
typedef __attribute__((ext_vector_type(4)))  float        v4f;

// ---------- bf16 helpers (round-to-nearest-even, finite inputs) ----------
__device__ __forceinline__ unsigned int f32_to_bf16_rne(float f) {
    unsigned int u = __builtin_bit_cast(unsigned int, f);
    u += 0x7FFFu + ((u >> 16) & 1u);
    return u >> 16;
}
__device__ __forceinline__ float bf16_bits_to_f32(unsigned int hi) {
    return __builtin_bit_cast(float, hi << 16);
}
__device__ __forceinline__ unsigned int pack2(float lo, float hi) {
    return f32_to_bf16_rne(lo) | (f32_to_bf16_rne(hi) << 16);
}
__device__ __forceinline__ v16bf frag_from_lds(const unsigned short* p0,
                                               const unsigned short* p1) {
    v4u a = *(const v4u*)p0;
    v4u b = *(const v4u*)p1;
    v8u r;
    r[0] = a[0]; r[1] = a[1]; r[2] = a[2]; r[3] = a[3];
    r[4] = b[0]; r[5] = b[1]; r[6] = b[2]; r[7] = b[3];
    return __builtin_bit_cast(v16bf, r);
}

// ---------- gfx1250 async global->LDS copy (ASYNCcnt-tracked DMA) ----------
__device__ __forceinline__ void async_load_b128(unsigned lds_byte_addr,
                                                unsigned gvoff,
                                                const void* gbase) {
    // GVS mode: mem_addr = SGPR64(gbase) + VGPR32(gvoff); LDS[vdst] <- 16B
    asm volatile("global_load_async_to_lds_b128 %0, %1, %2"
                 :: "v"(lds_byte_addr), "v"(gvoff), "s"(gbase) : "memory");
}
__device__ __forceinline__ void wait_async0() {
    asm volatile("s_wait_asynccnt 0x0" ::: "memory");
}

// =====================================================================
// Pre-pass A: W (1024x1023 f32) -> WhT/WlT (1024x1024 bf16, transposed,
// hi/lo split, col 1023 zero-padded). LDS-tiled 64x64 transpose.
// =====================================================================
__global__ __launch_bounds__(256) void prep_w_split_T(
    const float* __restrict__ W,
    unsigned short* __restrict__ WhT, unsigned short* __restrict__ WlT)
{
    __shared__ float tile[64][65];
    const int k0 = blockIdx.x * 64, n0 = blockIdx.y * 64;
    const int tid = threadIdx.x;
#pragma unroll
    for (int i = 0; i < 16; ++i) {
        int id = tid + i * 256;
        int r = id >> 6, c = id & 63;          // read coalesced over n
        int gn = n0 + c;
        tile[r][c] = (gn < 1023) ? W[(size_t)(k0 + r) * 1023 + gn] : 0.0f;
    }
    __syncthreads();
#pragma unroll
    for (int i = 0; i < 16; ++i) {
        int id = tid + i * 256;
        int n = id >> 6, k = id & 63;          // write coalesced over k
        float v = tile[k][n];
        unsigned int hb = f32_to_bf16_rne(v);
        float lo = v - bf16_bits_to_f32(hb);
        size_t o = (size_t)(n0 + n) * 1024 + k0 + k;
        WhT[o] = (unsigned short)hb;
        WlT[o] = (unsigned short)f32_to_bf16_rne(lo);
    }
}

// =====================================================================
// Pre-pass B: leaf_logits (1024x64 f32) -> Lb, pre-swizzled into the exact
// per-lane WMMA B-fragment layout: Lb[((ks*4+nt)*32+lane)*8 + v] =
// pack2(L[K][col], L[K+1][col]), K = ks*32+16*(lane/16)+2v, col = nt*16+lane%16
// =====================================================================
__global__ __launch_bounds__(256) void prep_leaf_swizzle(
    const float* __restrict__ L, unsigned int* __restrict__ Lb)
{
    int id = blockIdx.x * 256 + threadIdx.x;   // 32768 uints total
    int v    = id & 7;
    int lane = (id >> 3) & 31;
    int nt   = (id >> 8) & 3;
    int ks   = id >> 10;
    int K   = ks * 32 + (lane >> 4) * 16 + 2 * v;
    int col = nt * 16 + (lane & 15);
    Lb[id] = pack2(L[(size_t)K * 64 + col], L[(size_t)(K + 1) * 64 + col]);
}

// =====================================================================
// Kernel 1: P = sigmoid(X @ W + b), split-bf16 WMMA GEMM.
// Macro tile 128x128, K step 32, 8 waves (4Mx2N), wave tile 32x64 (2x4 WMMA),
// 3 WMMA per tile (hi*hi + hi*lo + lo*hi). Double-buffered LDS; B tiles
// staged with global_load_async_to_lds_b128 from pre-split WhT/WlT; A tiles
// register-prefetched + converted while WMMA runs.
// =====================================================================
#define K1_LDA 40            // padded ushort row stride (80B)
#define K1_SEG 5120          // 128*40 ushorts per sub-array
#define K1_BUF (4*K1_SEG)    // Ah|Al|Bh|Bl per buffer (40KB)

__global__ __launch_bounds__(256) void sdt_gemm_sigmoid(
    const float* __restrict__ X,
    const unsigned short* __restrict__ WhT, const unsigned short* __restrict__ WlT,
    const float* __restrict__ bias, float* __restrict__ P)
{
    extern __shared__ unsigned short lds1[];   // 2 * 40KB, dynamic

    const int D = 1024, NI = 1023, NP = 1024;
    const int tid  = threadIdx.x;
    const int lane = tid & 31, wave = tid >> 5;
    const int wm = wave & 3, wn = wave >> 2;
    const int half = lane >> 4, nlo = lane & 15;
    const int row0 = blockIdx.x * 128;
    const int col0 = blockIdx.y * 128;

    v8f acc[2][4];
#pragma unroll
    for (int mt = 0; mt < 2; ++mt)
#pragma unroll
        for (int nt = 0; nt < 4; ++nt) acc[mt][nt] = (v8f)(0.0f);

    // ---- helpers ----
    auto stageB_async = [&](int k0, int buf) {
        unsigned bh = (unsigned)(uintptr_t)(lds1 + buf * K1_BUF + 2 * K1_SEG);
        unsigned bl = (unsigned)(uintptr_t)(lds1 + buf * K1_BUF + 3 * K1_SEG);
#pragma unroll
        for (int i = 0; i < 2; ++i) {
            int id = tid + i * 256;            // 512 x 16B chunks
            int n = id >> 2, c16 = id & 3;
            unsigned goff = (unsigned)((((col0 + n) << 10) + k0) * 2 + c16 * 16);
            unsigned loff = (unsigned)(n * (K1_LDA * 2) + c16 * 16);
            async_load_b128(bh + loff, goff, WhT);
            async_load_b128(bl + loff, goff, WlT);
        }
    };
    auto loadA = [&](int k0, v4f* regs) {
#pragma unroll
        for (int i = 0; i < 4; ++i) {
            int id = tid + i * 256;
            int r = id >> 3, c4 = (id & 7) * 4;
            regs[i] = *(const v4f*)(X + (size_t)(row0 + r) * D + k0 + c4);
        }
    };
    auto storeA = [&](const v4f* regs, int buf) {
        unsigned short* Ah = lds1 + buf * K1_BUF;
        unsigned short* Al = Ah + K1_SEG;
#pragma unroll
        for (int i = 0; i < 4; ++i) {
            int id = tid + i * 256;
            int r = id >> 3, c4 = (id & 7) * 4;
#pragma unroll
            for (int j = 0; j < 4; ++j) {
                unsigned int hb = f32_to_bf16_rne(regs[i][j]);
                float lo = regs[i][j] - bf16_bits_to_f32(hb);
                Ah[r * K1_LDA + c4 + j] = (unsigned short)hb;
                Al[r * K1_LDA + c4 + j] = (unsigned short)f32_to_bf16_rne(lo);
            }
        }
    };
    auto compute = [&](int buf) {
        const unsigned short* Ah = lds1 + buf * K1_BUF;
        const unsigned short* Al = Ah + K1_SEG;
        const unsigned short* Bh = Ah + 2 * K1_SEG;
        const unsigned short* Bl = Ah + 3 * K1_SEG;
        v16bf ahf[2], alf[2];
#pragma unroll
        for (int mt = 0; mt < 2; ++mt) {
            const unsigned short* pa = Ah + (wm * 32 + mt * 16 + nlo) * K1_LDA + half * 8;
            const unsigned short* pl = Al + (wm * 32 + mt * 16 + nlo) * K1_LDA + half * 8;
            ahf[mt] = frag_from_lds(pa, pa + 16);
            alf[mt] = frag_from_lds(pl, pl + 16);
        }
#pragma unroll
        for (int nt = 0; nt < 4; ++nt) {
            const unsigned short* pb = Bh + (wn * 64 + nt * 16 + nlo) * K1_LDA + half * 8;
            const unsigned short* pL = Bl + (wn * 64 + nt * 16 + nlo) * K1_LDA + half * 8;
            v16bf bhf = frag_from_lds(pb, pb + 16);
            v16bf blf = frag_from_lds(pL, pL + 16);
#pragma unroll
            for (int mt = 0; mt < 2; ++mt) {
                acc[mt][nt] = __builtin_amdgcn_wmma_f32_16x16x32_bf16(
                    false, ahf[mt], false, bhf, (short)0, acc[mt][nt], false, false);
                acc[mt][nt] = __builtin_amdgcn_wmma_f32_16x16x32_bf16(
                    false, alf[mt], false, bhf, (short)0, acc[mt][nt], false, false);
                acc[mt][nt] = __builtin_amdgcn_wmma_f32_16x16x32_bf16(
                    false, ahf[mt], false, blf, (short)0, acc[mt][nt], false, false);
            }
        }
    };

    // ---- prologue: stage tile 0 ----
    v4f areg[4];
    stageB_async(0, 0);
    loadA(0, areg);
    storeA(areg, 0);
    wait_async0();
    __syncthreads();

    // ---- double-buffered main loop: async(t+1) overlaps WMMA(t) ----
    for (int t = 0; t < 32; ++t) {
        int cur = t & 1, nxt = cur ^ 1;
        if (t + 1 < 32) {
            stageB_async((t + 1) * 32, nxt);
            loadA((t + 1) * 32, areg);
        }
        compute(cur);
        if (t + 1 < 32) {
            storeA(areg, nxt);
            wait_async0();
        }
        __syncthreads();
    }

    // ---- epilogue: bias + sigmoid ----
#pragma unroll
    for (int nt = 0; nt < 4; ++nt) {
        int gcol = col0 + wn * 64 + nt * 16 + nlo;
        float bv = (gcol < NI) ? bias[gcol] : 0.0f;
#pragma unroll
        for (int mt = 0; mt < 2; ++mt) {
#pragma unroll
            for (int r = 0; r < 8; ++r) {
                int grow = row0 + wm * 32 + mt * 16 + r + half * 8;
                float s = acc[mt][nt][r] + bv;
                P[(size_t)grow * NP + gcol] = 1.0f / (1.0f + __expf(-s));
            }
        }
    }
}

// =====================================================================
// Kernel 2: in-place tree expansion in LDS + logits = leaf_prob @ leaf_logits
// 4 waves x 16 rows; B fragments come pre-swizzled from Lb (2x b128 loads).
// =====================================================================
#define K2_RS 1028   // padded float row stride: 16B aligned, conflict-free

__global__ __launch_bounds__(128) void sdt_tree_logits(
    const float* __restrict__ P, const unsigned int* __restrict__ Lb,
    float* __restrict__ out)
{
    extern __shared__ float smem[];
    const int NP = 1024, NA = 64;
    const int lane = threadIdx.x & 31, wave = threadIdx.x >> 5;
    const int half = lane >> 4, nlo = lane & 15;
    float* pr = smem + wave * 16 * K2_RS;
    const int rowbase = blockIdx.x * 64 + wave * 16;

    // level 0 -> level 1
    if (lane < 16) {
        float p0 = P[(size_t)(rowbase + lane) * NP];
        pr[lane * K2_RS + 0] = 1.0f - p0;
        pr[lane * K2_RS + 1] = p0;
    }
    __builtin_amdgcn_wave_barrier();

    // expand levels 1..9 in place (descending chunks; same-wave LDS in-order)
    for (int d = 1; d <= 9; ++d) {
        int nodes = 1 << d, first = nodes - 1;
        for (int r = 0; r < 16; ++r) {
            float* rowp = pr + r * K2_RS;
            const float* prow = P + (size_t)(rowbase + r) * NP;
            if (nodes <= 32) {
                int i = lane;
                bool act = (i < nodes);
                float c = 0.0f, pv = 0.0f;
                if (act) { c = rowp[i]; pv = prow[first + i]; }
                __builtin_amdgcn_wave_barrier();
                if (act) {
                    rowp[2 * i]     = c * (1.0f - pv);
                    rowp[2 * i + 1] = c * pv;
                }
            } else {
                for (int cb = nodes - 32; cb >= 0; cb -= 32) {
                    int i = cb + lane;
                    float c  = rowp[i];
                    float pv = prow[first + i];
                    __builtin_amdgcn_wave_barrier();
                    rowp[2 * i]     = c * (1.0f - pv);
                    rowp[2 * i + 1] = c * pv;
                }
            }
        }
        __builtin_amdgcn_wave_barrier();
    }

    // logits = leaf_prob (16x1024) @ leaf_logits (1024x64) via bf16 WMMA
    const float* myrow = pr + nlo * K2_RS;
    v8f acc[4];
#pragma unroll
    for (int nt = 0; nt < 4; ++nt) acc[nt] = (v8f)(0.0f);

    for (int ks = 0; ks < 32; ++ks) {
        int kb = ks * 32;
        const v4f* c0 = (const v4f*)(myrow + kb + half * 16);
        const v4f* c1 = (const v4f*)(myrow + kb + half * 16 + 8);
        v4f x0 = c0[0], x1 = c0[1], x2 = c1[0], x3 = c1[1];
        v8u au;
        au[0] = pack2(x0[0], x0[1]); au[1] = pack2(x0[2], x0[3]);
        au[2] = pack2(x1[0], x1[1]); au[3] = pack2(x1[2], x1[3]);
        au[4] = pack2(x2[0], x2[1]); au[5] = pack2(x2[2], x2[3]);
        au[6] = pack2(x3[0], x3[1]); au[7] = pack2(x3[2], x3[3]);
        v16bf afrag = __builtin_bit_cast(v16bf, au);
#pragma unroll
        for (int nt = 0; nt < 4; ++nt) {
            const v4u* bp = (const v4u*)(Lb + ((((size_t)ks * 4 + nt) * 32 + lane) << 3));
            v4u b0 = bp[0], b1 = bp[1];
            v8u bu;
            bu[0] = b0[0]; bu[1] = b0[1]; bu[2] = b0[2]; bu[3] = b0[3];
            bu[4] = b1[0]; bu[5] = b1[1]; bu[6] = b1[2]; bu[7] = b1[3];
            v16bf bfrag = __builtin_bit_cast(v16bf, bu);
            acc[nt] = __builtin_amdgcn_wmma_f32_16x16x32_bf16(
                false, afrag, false, bfrag, (short)0, acc[nt], false, false);
        }
    }

#pragma unroll
    for (int nt = 0; nt < 4; ++nt)
#pragma unroll
        for (int r = 0; r < 8; ++r) {
            int grow = rowbase + r + half * 8;
            out[(size_t)grow * NA + nt * 16 + nlo] = acc[nt][r];
        }
}

// =====================================================================
extern "C" void kernel_launch(void* const* d_in, const int* in_sizes, int n_in,
                              void* d_out, int out_size, void* d_ws, size_t ws_size,
                              hipStream_t stream) {
    (void)n_in; (void)out_size; (void)ws_size;
    const float* X  = (const float*)d_in[0];   // [B,1024]
    const float* W  = (const float*)d_in[1];   // [1024,1023]
    const float* b  = (const float*)d_in[2];   // [1023]
    const float* L  = (const float*)d_in[3];   // [1024,64]
    float* out = (float*)d_out;                // [B,64]

    int B = in_sizes[0] / 1024;                // 8192

    // workspace layout: P | WhT | WlT | Lb   (~36.2 MB total)
    char* wsb = (char*)d_ws;
    float* P = (float*)wsb;
    size_t off = (size_t)B * 1024 * sizeof(float);
    unsigned short* WhT = (unsigned short*)(wsb + off);           // 2 MB
    unsigned short* WlT = (unsigned short*)(wsb + off + 2097152); // 2 MB
    unsigned int*   Lb  = (unsigned int*)(wsb + off + 4194304);   // 128 KB

    prep_w_split_T<<<dim3(16, 16), 256, 0, stream>>>(W, WhT, WlT);
    prep_leaf_swizzle<<<128, 256, 0, stream>>>(L, Lb);

    dim3 g1(B / 128, 8);
    size_t lds1 = (size_t)2 * K1_BUF * sizeof(unsigned short);    // 81920 B
    sdt_gemm_sigmoid<<<g1, 256, lds1, stream>>>(X, WhT, WlT, b, P);

    size_t lds2 = (size_t)4 * 16 * K2_RS * sizeof(float);         // 263168 B
    sdt_tree_logits<<<B / 64, 128, lds2, stream>>>(P, Lb, out);
}
